// RVQ_58463094833503
// MI455X (gfx1250) — matrix operand order repaired and checked by reading, not compiled, not run on hardware
//
#include <hip/hip_runtime.h>

// ---------------------------------------------------------------------------
// RVQ codec for MI455X (gfx1250, wave32, WMMA).
//
// Roofline: 129 MB HBM traffic (~5.5us @ 23.3 TB/s) vs 8.6 GFLOP -> needs
// >1.4 PFLOP/s to be memory bound. f32 WMMA (16x16x4) is too slow; bf16 WMMA
// (v_wmma_f32_16x16x32_bf16, f32 accumulate) keeps this memory-bound.
// Weights (~65KB) are converted to bf16 once per launch into d_ws and stay
// L2-resident; the mel stream and recon stream go through LDS for fully
// coalesced 128B global transactions.
// ---------------------------------------------------------------------------

#define B_   16
#define T_   8192
#define MEL_ 128
#define D_   64
#define K_   128
#define C_   2
#define N_   (B_ * T_)       // 131072 rows

// d_ws layout (bytes)
#define WS_WIN  0            // bf16 [64][128]   = 16384 B
#define WS_WOUT 16384        // bf16 [128][64]   = 16384 B
#define WS_CB   32768        // bf16 [2][128][64]= 32768 B
#define WS_CSQ  65536        // f32  [2][128]    = 1024 B

typedef float  v8f   __attribute__((ext_vector_type(8)));
typedef __bf16 v16bf __attribute__((ext_vector_type(16)));

// POD vector types (HIP's float4/uint4 have ctors -> awkward in unions)
struct __align__(16) U4 { unsigned int a, b, c, d; };
struct __align__(8)  U2 { unsigned int a, b; };
struct __align__(16) F4 { float x, y, z, w; };

// --- WMMA fragment gathers (ISA 7.12.2 layouts, wave32) --------------------
// A (16x32 bf16, MxK): lane m=l&15 is row M; half h=l>>4 selects K sub-range.
//   VGPR0..7 hold K = k + 16h .. +15 (contiguous, matches frag_b)
__device__ __forceinline__ v16bf frag_a(const __bf16* p, int stride, int k,
                                        int m, int h) {
  union { v16bf v; U4 q[2]; } u;
  const char* b = (const char*)(p + m * stride + k + h * 16);
  u.q[0] = *(const U4*)(b);
  u.q[1] = *(const U4*)(b + 16);   // +8 bf16 elements
  return u.v;
}
// B (32x16 bf16, KxN): lane m is column N; lane half h selects K=16h..16h+15,
// i.e. 32 contiguous bytes from the source row (source is stored N-major,
// K-contiguous, which is exactly how W (d,k) / codebook (j,d) rows lie).
__device__ __forceinline__ v16bf frag_b(const __bf16* p, int stride, int k,
                                        int m, int h) {
  union { v16bf v; U4 q[2]; } u;
  const U4* b = (const U4*)(p + m * stride + k + h * 16);
  u.q[0] = b[0];
  u.q[1] = b[1];
  return u.v;
}

#define WMMA_BF16(A, Bm, Cacc) \
  __builtin_amdgcn_wmma_f32_16x16x32_bf16(false, (A), false, (Bm), (short)0, \
                                          (Cacc), false, false)

// ---------------------------------------------------------------------------
// Prep: convert weights/codebooks to bf16 in d_ws, precompute ||code||^2 in
// f32 from the original f32 codebooks (argmin bias term stays accurate).
// ---------------------------------------------------------------------------
__global__ void rvq_prep_58463094833503(const float* __restrict__ win,
                                        const float* __restrict__ wout,
                                        const float* __restrict__ cb,
                                        unsigned char* __restrict__ ws) {
  __bf16* winb  = (__bf16*)(ws + WS_WIN);
  __bf16* woutb = (__bf16*)(ws + WS_WOUT);
  __bf16* cbb   = (__bf16*)(ws + WS_CB);
  float*  csq   = (float*)(ws + WS_CSQ);
  int t0 = blockIdx.x * blockDim.x + threadIdx.x;
  int nt = gridDim.x * blockDim.x;
  for (int i = t0; i < D_ * MEL_; i += nt) winb[i]  = (__bf16)win[i];
  for (int i = t0; i < MEL_ * D_; i += nt) woutb[i] = (__bf16)wout[i];
  for (int i = t0; i < C_ * K_ * D_; i += nt) cbb[i] = (__bf16)cb[i];
  for (int i = t0; i < C_ * K_; i += nt) {
    float s = 0.f;
    for (int d = 0; d < D_; ++d) { float v = cb[i * D_ + d]; s += v * v; }
    csq[i] = s;
  }
}

// ---------------------------------------------------------------------------
// Main kernel: 128 threads = 4 wave32; each wave owns a 16-row tile.
// ---------------------------------------------------------------------------
#define NW 4                     // waves per block
#define XS 72                    // bf16 row stride in LDS (pad: no bank dup)
#define FS 68                    // f32 row stride for recon staging

__global__ __launch_bounds__(128)
void rvq_main_58463094833503(const float* __restrict__ x,
                             const float* __restrict__ b_in,
                             const float* __restrict__ b_out,
                             const unsigned char* __restrict__ ws,
                             float* __restrict__ out) {
  // per-wave staging: phase1 uses it as bf16 [16][72] (2304B),
  // phase3 reuses it as f32 [16][68] (4352B) recon staging.
  __shared__ __align__(16) unsigned char s_buf[NW][16 * FS * 4];
  __shared__ __align__(16) __bf16        s_rq[NW][16 * XS];  // residual/q tile
  __shared__ int                         s_best[NW][C_][16];

  const __bf16* winb  = (const __bf16*)(ws + WS_WIN);
  const __bf16* woutb = (const __bf16*)(ws + WS_WOUT);
  const __bf16* cbb   = (const __bf16*)(ws + WS_CB);
  const float*  csq   = (const float*)(ws + WS_CSQ);

  const int tid = threadIdx.x;
  const int wv  = tid >> 5;
  const int l   = tid & 31;
  const int m   = l & 15;       // row/col within a 16-wide fragment
  const int h   = l >> 4;       // lane half
  const long rowBase = ((long)blockIdx.x * NW + wv) * 16;

  float* outIdx = out;                       // (N, C) indices as f32 values
  float* outRec = out + (size_t)N_ * C_;     // (N, MEL) recon

  // ======================= Phase 1: z = x @ Win^T + b ======================
  v8f accz[4];
  for (int dt = 0; dt < 4; ++dt) accz[dt] = (v8f){};

  for (int kh = 0; kh < 2; ++kh) {          // two 64-wide K halves of MEL
    __bf16* sx = (__bf16*)s_buf[wv];
    // coalesced f32 load of x[rows][kh*64..+64) -> bf16 LDS tile
    for (int i = 0; i < 8; ++i) {
      int t  = i * 32 + l;                  // 0..255 over 16 rows x 16 f4
      int r  = t >> 4, c4 = t & 15;
      F4 f = *(const F4*)(x + (rowBase + r) * MEL_ + kh * 64 + c4 * 4);
      union { __bf16 hh[4]; U2 u; } pk;
      pk.hh[0] = (__bf16)f.x; pk.hh[1] = (__bf16)f.y;
      pk.hh[2] = (__bf16)f.z; pk.hh[3] = (__bf16)f.w;
      *(U2*)((char*)sx + r * (XS * 2) + c4 * 8) = pk.u;
    }
    __syncthreads();
    for (int k0 = 0; k0 < 64; k0 += 32) {
      v16bf a = frag_a(sx, XS, k0, m, h);
      int kg = kh * 64 + k0;
      for (int dt = 0; dt < 4; ++dt) {
        v16bf b = frag_b(winb + dt * 16 * MEL_, MEL_, kg, m, h);
        accz[dt] = WMMA_BF16(a, b, accz[dt]);
      }
    }
    __syncthreads();
  }
  // bias add + spill z to residual tile (C layout: slot s -> row s+8h)
  for (int dt = 0; dt < 4; ++dt) {
    float bi = b_in[dt * 16 + m];
    for (int s = 0; s < 8; ++s) {
      int r = s + 8 * h;
      s_rq[wv][r * XS + dt * 16 + m] = (__bf16)(accz[dt][s] + bi);
    }
  }
  __syncthreads();

  // ================= Phase 2: two sequential codebook stages ===============
  // dist argmin == argmin_j (||c_j||^2 - 2 r.c_j): row norm dropped.
  for (int c = 0; c < C_; ++c) {
    const __bf16* cbc = cbb + c * K_ * D_;
    v8f S[8];
    for (int jt = 0; jt < 8; ++jt) S[jt] = (v8f){};
    for (int k0 = 0; k0 < 64; k0 += 32) {
      v16bf a = frag_a(s_rq[wv], XS, k0, m, h);
      for (int jt = 0; jt < 8; ++jt) {
        v16bf b = frag_b(cbc + jt * 16 * D_, D_, k0, m, h);
        S[jt] = WMMA_BF16(a, b, S[jt]);
      }
    }
    // per-lane argmin over the 8 j-tiles (j = jt*16 + m), 8 row slots each
    float bv[8]; int bi_[8];
    for (int s = 0; s < 8; ++s) { bv[s] = 3.4e38f; bi_[s] = 0; }
    for (int jt = 0; jt < 8; ++jt) {
      int   j  = jt * 16 + m;
      float cq = csq[c * K_ + j];
      for (int s = 0; s < 8; ++s) {
        float val = cq - 2.0f * S[jt][s];
        if (val < bv[s]) { bv[s] = val; bi_[s] = j; }   // ascending j: first-min
      }
    }
    // min-reduce across the 16 lanes of each half (stays within the half)
    for (int off = 8; off >= 1; off >>= 1) {
      for (int s = 0; s < 8; ++s) {
        float ov = __shfl_xor(bv[s], off);
        int   oi = __shfl_xor(bi_[s], off);
        if (ov < bv[s] || (ov == bv[s] && oi < bi_[s])) { bv[s] = ov; bi_[s] = oi; }
      }
    }
    if (m == 0) {                     // lanes 0 (rows 0-7) and 16 (rows 8-15)
      for (int s = 0; s < 8; ++s) {
        int r = s + 8 * h;
        s_best[wv][c][r] = bi_[s];
        outIdx[(rowBase + r) * C_ + c] = (float)bi_[s];
      }
    }
    __syncthreads();
    // residual update (c==0) or q = C0[i0]+C1[i1] (c==1); 2 lanes per row.
    {
      int r = l >> 1, db = (l & 1) * 32;
      if (c == 0) {
        int i0 = s_best[wv][0][r];
        for (int i = 0; i < 4; ++i) {
          union { U4 q; __bf16 hh[8]; } rv, cv;
          char* rp = (char*)s_rq[wv] + r * (XS * 2) + (db + i * 8) * 2;
          rv.q = *(U4*)rp;
          cv.q = *(const U4*)((const char*)cbb + i0 * (D_ * 2) + (db + i * 8) * 2);
          for (int e = 0; e < 8; ++e)
            rv.hh[e] = (__bf16)((float)rv.hh[e] - (float)cv.hh[e]);
          *(U4*)rp = rv.q;
        }
      } else {
        int i0 = s_best[wv][0][r], i1 = s_best[wv][1][r];
        for (int i = 0; i < 4; ++i) {
          union { U4 q; __bf16 hh[8]; } c0v, c1v, qv;
          c0v.q = *(const U4*)((const char*)cbb + i0 * (D_ * 2) + (db + i * 8) * 2);
          c1v.q = *(const U4*)((const char*)cbb + (K_ + i1) * (D_ * 2) + (db + i * 8) * 2);
          for (int e = 0; e < 8; ++e)
            qv.hh[e] = (__bf16)((float)c0v.hh[e] + (float)c1v.hh[e]);
          *(U4*)((char*)s_rq[wv] + r * (XS * 2) + (db + i * 8) * 2) = qv.q;
        }
      }
    }
    __syncthreads();
  }

  // ================= Phase 3: recon = q @ Wout^T + b_out ===================
  for (int mh = 0; mh < 2; ++mh) {          // two 64-wide halves of MEL out
    v8f ar[4];
    for (int mt = 0; mt < 4; ++mt) ar[mt] = (v8f){};
    for (int k0 = 0; k0 < 64; k0 += 32) {
      v16bf a = frag_a(s_rq[wv], XS, k0, m, h);
      for (int mt = 0; mt < 4; ++mt) {
        v16bf b = frag_b(woutb + (mh * 64 + mt * 16) * D_, D_, k0, m, h);
        ar[mt] = WMMA_BF16(a, b, ar[mt]);
      }
    }
    float* sf = (float*)s_buf[wv];          // reuse staging as f32 [16][68]
    for (int mt = 0; mt < 4; ++mt) {
      float bo = b_out[mh * 64 + mt * 16 + m];
      for (int s = 0; s < 8; ++s) {
        int r = s + 8 * h;
        sf[r * FS + mt * 16 + m] = ar[mt][s] + bo;
      }
    }
    __syncthreads();
    // coalesced float4 stores: 16 consecutive float4 per row per half-wave
    for (int i = 0; i < 8; ++i) {
      int t = i * 32 + l;
      int r = t >> 4, c4 = t & 15;
      F4 v = *(const F4*)(sf + r * FS + c4 * 4);
      *(F4*)(outRec + (rowBase + r) * MEL_ + mh * 64 + c4 * 4) = v;
    }
    __syncthreads();
  }
}

extern "C" void kernel_launch(void* const* d_in, const int* in_sizes, int n_in,
                              void* d_out, int out_size, void* d_ws, size_t ws_size,
                              hipStream_t stream) {
  (void)in_sizes; (void)n_in; (void)out_size; (void)ws_size;
  const float* x    = (const float*)d_in[0];   // mel_frame (B,T,MEL)
  const float* win  = (const float*)d_in[1];   // proj_in_w (D,MEL)
  const float* bin  = (const float*)d_in[2];   // proj_in_b (D)
  const float* cb   = (const float*)d_in[3];   // codebooks (C,K,D)
  const float* wout = (const float*)d_in[4];   // proj_out_w (MEL,D)
  const float* bout = (const float*)d_in[5];   // proj_out_b (MEL)
  unsigned char* ws = (unsigned char*)d_ws;
  float* out = (float*)d_out;

  rvq_prep_58463094833503<<<64, 256, 0, stream>>>(win, wout, cb, ws);
  // 131072 rows / (4 waves * 16 rows) = 2048 blocks of 128 threads
  rvq_main_58463094833503<<<N_ / (NW * 16), 128, 0, stream>>>(x, bin, bout, ws, out);
}
